// GQAttention_74586402063062
// MI455X (gfx1250) — matrix operand order repaired and checked by reading, not compiled, not run on hardware
//
#include <hip/hip_runtime.h>
#include <hip/hip_bf16.h>

// ---------------------------------------------------------------------------
// CDNA5 (gfx1250) wave32 WMMA implementation of a GQA attention block.
//   q  = x @ wq            (bf16 WMMA, f32 accum)   -> ws
//   kv = x @ wkv           (bf16 WMMA, f32 accum)   -> ws
//   RoPE on q and k        (f32 VALU, in place on bf16 ws)
//   flash attention (causal, GQA 16q/4kv, D=64)     -> ws (bf16)
//   out = y @ cproj        (bf16 WMMA, f32 out)     -> d_out
// ---------------------------------------------------------------------------

typedef __attribute__((ext_vector_type(16))) __bf16 v16bf;
typedef __attribute__((ext_vector_type(8)))  __bf16 v8bf;
typedef __attribute__((ext_vector_type(8)))  float  v8f;

union ABf { v16bf v; struct { v8bf lo; v8bf hi; } h; };

__device__ __forceinline__ __bf16 f2bf(float f) {
  unsigned int u = __builtin_bit_cast(unsigned int, f);
  unsigned int r = (u + 0x7FFFu + ((u >> 16) & 1u)) >> 16;
  return __builtin_bit_cast(__bf16, (unsigned short)r);
}
__device__ __forceinline__ float bf2f(__bf16 b) {
  unsigned int u = (unsigned int)__builtin_bit_cast(unsigned short, b) << 16;
  return __builtin_bit_cast(float, u);
}
__device__ __forceinline__ v8f wmma_bf16(v16bf a, v16bf b, v8f c) {
  return __builtin_amdgcn_wmma_f32_16x16x32_bf16(
      /*neg_a=*/false, a, /*neg_b=*/false, b,
      /*c_mod=*/(short)0, c, /*reuse_a=*/false, /*reuse_b=*/false);
}

// ---- CDNA5 async global->LDS DMA (tracked by ASYNCcnt) ----
__device__ __forceinline__ void async_load_b128(const void* g, void* l) {
  unsigned lofs = (unsigned)(unsigned long long)l;  // low 32b = LDS offset
  asm volatile("global_load_async_to_lds_b128 %0, %1, off"
               :: "v"(lofs), "v"((unsigned long long)g) : "memory");
}
__device__ __forceinline__ void wait_async0() {
  asm volatile("s_wait_asynccnt 0x0" ::: "memory");
}

// ---- DPP16 cross-lane reductions over 16-lane rows (no LDS round trip) ----
// 0xB1 = quad_perm xor1, 0x4E = quad_perm xor2,
// 0x141 = ROW_HALF_MIRROR, 0x140 = ROW_MIRROR
template <int CTRL>
__device__ __forceinline__ float dpp_bcast(float x) {
  int r = __builtin_amdgcn_update_dpp(0, __builtin_bit_cast(int, x),
                                      CTRL, 0xF, 0xF, true);
  return __builtin_bit_cast(float, r);
}
__device__ __forceinline__ float red_max16(float v) {
  v = fmaxf(v, dpp_bcast<0xB1>(v));
  v = fmaxf(v, dpp_bcast<0x4E>(v));
  v = fmaxf(v, dpp_bcast<0x141>(v));
  v = fmaxf(v, dpp_bcast<0x140>(v));
  return v;
}
__device__ __forceinline__ float red_sum16(float v) {
  v += dpp_bcast<0xB1>(v);
  v += dpp_bcast<0x4E>(v);
  v += dpp_bcast<0x141>(v);
  v += dpp_bcast<0x140>(v);
  return v;
}

// ---------------------------------------------------------------------------
// GEMM:  Out[M,N] = A[M,K] * W[K,N];  A is f32 or bf16, W is f32.
// Block: 256 thr (8 waves), tile 128x64, K-step 32.  Waves 4x2, each 32x32.
// Double-buffered LDS: stage tile t+1 into registers while WMMAs run on
// tile t, one barrier per K-step.
// ---------------------------------------------------------------------------
#define BM 128
#define BN 64
#define BK 32

template <bool A_BF16, bool OUT_BF16>
__global__ __launch_bounds__(256)
void gemm_wmma(const void* __restrict__ Aptr, const float* __restrict__ W,
               void* __restrict__ Out, int M, int N, int K) {
  __shared__ __bf16 ldsA[2][BM * BK];  // [m][k] row-major
  __shared__ __bf16 ldsB[2][BN * BK];  // [n][k] (transposed tile)

  const int tid  = threadIdx.x;
  const int lane = tid & 31;
  const int wid  = tid >> 5;
  const int wm   = wid >> 1;        // 0..3 : 32 rows
  const int wn   = wid & 1;         // 0..1 : 32 cols
  const int half = lane >> 4;       // 0/1
  const int l16  = lane & 15;
  const int m0   = blockIdx.y * BM;
  const int n0   = blockIdx.x * BN;

  const int ar = tid >> 3;          // 0..31  (A stage row base)
  const int ac = (tid & 7) * 4;     // 0..28  (A stage col)
  const int bk = tid & 31;          // 0..31  (B stage k)
  const int bn = (tid >> 5) * 8;    // 0..56  (B stage n base)

  float4 raf[4];                    // staged A (f32 path)
  uint2  rab[4];                    // staged A (bf16 path)
  float4 rb0, rb1;                  // staged B

  auto gload = [&](int k0) {
#pragma unroll
    for (int p = 0; p < 4; ++p) {
      if (A_BF16)
        rab[p] = *(const uint2*)((const __bf16*)Aptr +
                                 (size_t)(m0 + ar + p * 32) * K + k0 + ac);
      else
        raf[p] = *(const float4*)((const float*)Aptr +
                                  (size_t)(m0 + ar + p * 32) * K + k0 + ac);
    }
    const float* g = W + (size_t)(k0 + bk) * N + n0 + bn;
    rb0 = *(const float4*)g;
    rb1 = *(const float4*)(g + 4);
    if (k0 + BK < K)
      __builtin_prefetch(W + (size_t)(k0 + BK + bk) * N + n0 + bn, 0, 0);
  };
  auto lstore = [&](int buf) {
#pragma unroll
    for (int p = 0; p < 4; ++p) {
      __bf16* d = &ldsA[buf][(ar + p * 32) * BK + ac];
      if (A_BF16) {
        *(uint2*)d = rab[p];
      } else {
        d[0] = f2bf(raf[p].x); d[1] = f2bf(raf[p].y);
        d[2] = f2bf(raf[p].z); d[3] = f2bf(raf[p].w);
      }
    }
    __bf16* db = &ldsB[buf][0];
    db[(bn + 0) * BK + bk] = f2bf(rb0.x);
    db[(bn + 1) * BK + bk] = f2bf(rb0.y);
    db[(bn + 2) * BK + bk] = f2bf(rb0.z);
    db[(bn + 3) * BK + bk] = f2bf(rb0.w);
    db[(bn + 4) * BK + bk] = f2bf(rb1.x);
    db[(bn + 5) * BK + bk] = f2bf(rb1.y);
    db[(bn + 6) * BK + bk] = f2bf(rb1.z);
    db[(bn + 7) * BK + bk] = f2bf(rb1.w);
  };

  v8f acc[2][2] = {};

  gload(0);
  lstore(0);
  __syncthreads();

  const int NT = K / BK;
  for (int t = 0; t < NT; ++t) {
    const int cur = t & 1;
    if (t + 1 < NT) gload((t + 1) * BK);   // overlap with WMMAs below

    // ---- fragments (ISA 7.12.2 layouts) ----
    ABf af[2];
#pragma unroll
    for (int mt = 0; mt < 2; ++mt) {
      const __bf16* p = &ldsA[cur][(wm * 32 + mt * 16 + l16) * BK + half * 8];
      af[mt].h.lo = *(const v8bf*)p;
      af[mt].h.hi = *(const v8bf*)(p + 16);
    }
    ABf bf[2];
#pragma unroll
    for (int nt = 0; nt < 2; ++nt) {
      const __bf16* p = &ldsB[cur][(wn * 32 + nt * 16 + l16) * BK + half * 8];
      bf[nt].h.lo = *(const v8bf*)p;
      bf[nt].h.hi = *(const v8bf*)(p + 16);
    }
#pragma unroll
    for (int mt = 0; mt < 2; ++mt)
#pragma unroll
      for (int nt = 0; nt < 2; ++nt)
        acc[mt][nt] = wmma_bf16(af[mt].v, bf[nt].v, acc[mt][nt]);

    if (t + 1 < NT) lstore(cur ^ 1);       // fill the other buffer
    __syncthreads();
  }

  // ---- epilogue ----
#pragma unroll
  for (int mt = 0; mt < 2; ++mt)
#pragma unroll
    for (int nt = 0; nt < 2; ++nt)
#pragma unroll
      for (int r = 0; r < 8; ++r) {
        int row = m0 + wm * 32 + mt * 16 + r + half * 8;
        int col = n0 + wn * 32 + nt * 16 + l16;
        float v = acc[mt][nt][r];
        if (OUT_BF16) ((__bf16*)Out)[(size_t)row * N + col] = f2bf(v);
        else          ((float*)Out)[(size_t)row * N + col] = v;
      }
}

// ---------------------------------------------------------------------------
// RoPE in-place on bf16 [B,S,NH,64]; pairs (i, i+32), i<32.
// ---------------------------------------------------------------------------
template <int NH, int STRIDE>
__global__ __launch_bounds__(256)
void rope_kernel(__bf16* __restrict__ p) {
  int idx = blockIdx.x * blockDim.x + threadIdx.x;   // B*2048*NH*32 threads
  int i    = idx & 31;
  int hd   = (idx >> 5) & (NH - 1);
  int pos  = (idx >> 5) / NH & 2047;
  int b    = (idx >> 5) / NH / 2048;
  __bf16* q = p + (size_t)(b * 2048 + pos) * STRIDE + hd * 64;
  // inv_freq = 10000^(-i/32) = exp(-i * ln(10000)/32)
  float inv = __expf(-(float)i * (9.2103403719761836f / 32.0f));
  float ang = (float)pos * inv;
  float sn, cs;
  __sincosf(ang, &sn, &cs);
  float x1 = bf2f(q[i]);
  float x2 = bf2f(q[i + 32]);
  q[i]      = f2bf(x1 * cs - x2 * sn);
  q[i + 32] = f2bf(x2 * cs + x1 * sn);
}

// ---------------------------------------------------------------------------
// Causal GQA flash attention. One WG per (b, h, 64-row q tile); 4 waves,
// each wave owns 16 q rows. K-blocks of 32 keys: K tile is DMA'd to LDS with
// async global->LDS (ASYNCcnt), V tile is transposed through VGPRs.
// Online softmax in f32 with DPP16 row reductions.
// ---------------------------------------------------------------------------
__global__ __launch_bounds__(128)
void attn_fa(const __bf16* __restrict__ qw, const __bf16* __restrict__ kvw,
             __bf16* __restrict__ yw) {
  const int b   = blockIdx.z;
  const int h   = blockIdx.y;
  const int q0  = blockIdx.x * 64;
  const int kvh = h >> 2;
  const int tid  = threadIdx.x;
  const int lane = tid & 31;
  const int wid  = tid >> 5;
  const int half = lane >> 4;
  const int l16  = lane & 15;
  const int qr0  = q0 + wid * 16;
  const float SCALE = 0.125f;  // 1/sqrt(64)

  __shared__ __bf16 ldsK[32 * 64];      // [key][d]
  __shared__ __bf16 ldsVt[64 * 32];     // [d][key]
  __shared__ __bf16 ldsP[4 * 16 * 32];  // per-wave P tile
  __bf16* pP = &ldsP[wid * 16 * 32];

  // Q fragments (A operand), pre-loaded once: rows qr0..qr0+15, d 0..63.
  ABf qf[2];
  {
    const __bf16* qp = qw + ((size_t)(b * 2048 + qr0 + l16) * 16 + h) * 64;
#pragma unroll
    for (int dc = 0; dc < 2; ++dc) {
      const __bf16* p = qp + dc * 32 + half * 8;
      qf[dc].h.lo = *(const v8bf*)p;
      qf[dc].h.hi = *(const v8bf*)(p + 16);
    }
  }

  float m_run[8], l_run[8];
  v8f accO[4] = {};
#pragma unroll
  for (int r = 0; r < 8; ++r) { m_run[r] = -1e30f; l_run[r] = 0.0f; }

  const int key = tid >> 2;          // 0..31
  const int d0  = (tid & 3) * 16;    // 0..48

  const int nkb = (q0 + 64) >> 5;
  for (int kb = 0; kb < nkb; ++kb) {
    // ---- stage K via async DMA, V via VGPR transpose ----
    {
      const __bf16* kp =
          kvw + ((size_t)(b * 2048 + kb * 32 + key) * 8 + kvh) * 64 + d0;
      async_load_b128(kp,     &ldsK[key * 64 + d0]);       // 8 bf16
      async_load_b128(kp + 8, &ldsK[key * 64 + d0 + 8]);   // 8 bf16
      const __bf16* vp = kp + 256;  // v starts 256 bf16 after k in kv row
      union { uint4 u[2]; __bf16 e[16]; } vv;
      vv.u[0] = *(const uint4*)vp;
      vv.u[1] = *(const uint4*)(vp + 8);
#pragma unroll
      for (int j = 0; j < 16; ++j) ldsVt[(d0 + j) * 32 + key] = vv.e[j];
      wait_async0();                 // K tile landed in LDS
    }
    __syncthreads();

    // ---- S = Q K^T  (two 16-key tiles) ----
    v8f s0 = {}, s1 = {};
#pragma unroll
    for (int dc = 0; dc < 2; ++dc) {
      ABf b0, b1;
      { const __bf16* p = &ldsK[l16 * 64 + dc * 32 + half * 8];
        b0.h.lo = *(const v8bf*)p; b0.h.hi = *(const v8bf*)(p + 16); }
      { const __bf16* p = &ldsK[(16 + l16) * 64 + dc * 32 + half * 8];
        b1.h.lo = *(const v8bf*)p; b1.h.hi = *(const v8bf*)(p + 16); }
      s0 = wmma_bf16(qf[dc].v, b0.v, s0);
      s1 = wmma_bf16(qf[dc].v, b1.v, s1);
    }

    // ---- scale + causal mask + online softmax (DPP reductions) ----
    const int c0 = kb * 32 + l16;
    const int c1 = c0 + 16;
#pragma unroll
    for (int r = 0; r < 8; ++r) {
      int rg = qr0 + r + half * 8;
      float v0 = (c0 <= rg) ? s0[r] * SCALE : -1e30f;
      float v1 = (c1 <= rg) ? s1[r] * SCALE : -1e30f;
      float mx = red_max16(fmaxf(v0, v1));
      float mnew  = fmaxf(m_run[r], mx);
      float alpha = __expf(m_run[r] - mnew);
      float p0 = __expf(v0 - mnew);
      float p1 = __expf(v1 - mnew);
      float sum = red_sum16(p0 + p1);
      l_run[r] = l_run[r] * alpha + sum;
      m_run[r] = mnew;
#pragma unroll
      for (int dt = 0; dt < 4; ++dt) accO[dt][r] *= alpha;
      pP[(r + half * 8) * 32 + l16]      = f2bf(p0);
      pP[(r + half * 8) * 32 + 16 + l16] = f2bf(p1);
    }

    // ---- O += P V  (A = P 16x32, B = V^T tiles) ----
    ABf pf;
    { const __bf16* p = &pP[l16 * 32 + half * 8];
      pf.h.lo = *(const v8bf*)p; pf.h.hi = *(const v8bf*)(p + 16); }
#pragma unroll
    for (int dt = 0; dt < 4; ++dt) {
      ABf vf;
      const __bf16* p = &ldsVt[(dt * 16 + l16) * 32 + half * 8];
      vf.h.lo = *(const v8bf*)p; vf.h.hi = *(const v8bf*)(p + 16);
      accO[dt] = wmma_bf16(pf.v, vf.v, accO[dt]);
    }
    __syncthreads();
  }

  // ---- epilogue: y = O / l ----
#pragma unroll
  for (int dt = 0; dt < 4; ++dt)
#pragma unroll
    for (int r = 0; r < 8; ++r) {
      int rg = qr0 + r + half * 8;
      int d  = dt * 16 + l16;
      float y = accO[dt][r] / l_run[r];
      yw[((size_t)(b * 2048 + rg) * 16 + h) * 64 + d] = f2bf(y);
    }
}

// ---------------------------------------------------------------------------
// Launch:  B=2 S=2048 C=1024 H=16 KVH=4 D=64;  M = B*S = 4096.
// ws: q(4096x1024 bf16) | kv(4096x512 bf16) | y(4096x1024 bf16) = 20 MB.
// ---------------------------------------------------------------------------
extern "C" void kernel_launch(void* const* d_in, const int* in_sizes, int n_in,
                              void* d_out, int out_size, void* d_ws, size_t ws_size,
                              hipStream_t stream) {
  const float* x     = (const float*)d_in[0];
  // d_in[1] = mask (all true) -> causal handled in-kernel
  const float* wq    = (const float*)d_in[2];
  const float* wkv   = (const float*)d_in[3];
  const float* cproj = (const float*)d_in[4];
  float* out = (float*)d_out;

  __bf16* qws  = (__bf16*)d_ws;                       // 4096*1024
  __bf16* kvws = qws + (size_t)4096 * 1024;           // 4096*512
  __bf16* yws  = kvws + (size_t)4096 * 512;           // 4096*1024

  dim3 blk(256);
  // q = x @ wq
  gemm_wmma<false, true><<<dim3(1024 / BN, 4096 / BM), blk, 0, stream>>>(
      x, wq, qws, 4096, 1024, 1024);
  // kv = x @ wkv
  gemm_wmma<false, true><<<dim3(512 / BN, 4096 / BM), blk, 0, stream>>>(
      x, wkv, kvws, 4096, 512, 1024);
  // RoPE on q and k
  rope_kernel<16, 1024><<<(2 * 2048 * 16 * 32) / 256, 256, 0, stream>>>(qws);
  rope_kernel<4, 512><<<(2 * 2048 * 4 * 32) / 256, 256, 0, stream>>>(kvws);
  // flash attention
  attn_fa<<<dim3(2048 / 64, 16, 2), 128, 0, stream>>>(qws, kvws, yws);
  // out = y @ cproj
  gemm_wmma<true, false><<<dim3(1024 / BN, 4096 / BM), blk, 0, stream>>>(
      yws, cproj, out, 4096, 1024, 1024);
}
